// S4wDecoder_78950088835615
// MI455X (gfx1250) — matrix operand order repaired and checked by reading, not compiled, not run on hardware
//
#include <hip/hip_runtime.h>
#include <hip/hip_bf16.h>

typedef unsigned short u16;
typedef unsigned int   u32;

typedef __attribute__((ext_vector_type(16))) __bf16 v16bf;
typedef __attribute__((ext_vector_type(8)))  __bf16 v8bf;
typedef __attribute__((ext_vector_type(8)))  float  v8f;
typedef __attribute__((ext_vector_type(4)))  int    v4i;

#define BN     32
#define LEN    2048
#define D_IN   64
#define HC     256
#define N2C    32
#define L1C    350
#define L2C    400
#define DOUT   1024
#define NLAY   2
#define LN_EPS 1e-5f

// ---- CDNA5 async global->LDS staging (ASYNCcnt path), feature-gated ----
#if defined(__HIP_DEVICE_COMPILE__) && __has_builtin(__builtin_amdgcn_global_load_async_to_lds_b128)
#define HAVE_ASYNC 1
#else
#define HAVE_ASYNC 0
#endif

// builtin signature (from clang diagnostic): (v4i as(1)*, v4i as(3)*, Ii offset, Ii cpol)
#define GAS1(p) ((__attribute__((address_space(1))) v4i*)(p))
#define LAS3(p) ((__attribute__((address_space(3))) v4i*)(p))

__device__ __forceinline__ void wait_async_all() {
#if HAVE_ASYNC
#if __has_builtin(__builtin_amdgcn_s_wait_asynccnt)
    __builtin_amdgcn_s_wait_asynccnt(0);
#else
    asm volatile("s_wait_asynccnt 0x0" ::: "memory");
#endif
#endif
}

// ---------- helpers ----------
__device__ __forceinline__ u16 f2bf(float f) {
    u32 u = __builtin_bit_cast(u32, f);
    u += 0x7FFFu + ((u >> 16) & 1u);   // round-to-nearest-even
    return (u16)(u >> 16);
}

// ---------- f32 -> bf16 (also used for x: (B,L,D_IN) stays layout-identical) ----------
__global__ void f2bf_kernel(const float* __restrict__ in, u16* __restrict__ out, int n) {
    int i = blockIdx.x * blockDim.x + threadIdx.x;
    if (i < n) out[i] = f2bf(in[i]);
}

// ---------- S4D parameter precompute:  par[i] = {Ct.re, Ct.im, w.re, w.im} ----------
__global__ void s4_params_kernel(const float* __restrict__ log_dt,
                                 const float* __restrict__ C_re,
                                 const float* __restrict__ C_im,
                                 const float* __restrict__ log_A_real,
                                 const float* __restrict__ A_imag,
                                 float4* __restrict__ par) {
    int i = blockIdx.x * blockDim.x + threadIdx.x;
    if (i >= NLAY * HC * N2C) return;
    int h     = (i / N2C) % HC;
    int layer = i / (HC * N2C);
    float dt = expf(log_dt[layer * HC + h]);
    float ar = -expf(log_A_real[i]);
    float ai = A_imag[i];
    float dr = ar * dt, di = ai * dt;
    float er = expf(dr);
    float wr = er * cosf(di), wi = er * sinf(di);     // w = exp(dt*A)
    float nr = wr - 1.0f, ni = wi;                    // exp(dtA) - 1
    float den = ar * ar + ai * ai;
    float qr = (nr * ar + ni * ai) / den;             // (w-1)/A
    float qi = (ni * ar - nr * ai) / den;
    float cr = C_re[i], ci = C_im[i];
    par[i] = make_float4(cr * qr - ci * qi, cr * qi + ci * qr, wr, wi);
}

// ---------- bf16 WMMA GEMM:  C[b](MxN) = A(MxK) * Bt[b](NxK)^T + bias ----------
// A row-major (M,K); Bt row-major (N,K)  -> both operands staged to LDS in
// WMMA-fragment order as pure 16-byte chunk moves (async-to-LDS capable).
#define TM 64
#define TN 64
#define TK 32
__global__ __launch_bounds__(256) void gemm_bf16_kernel(
    const u16*  __restrict__ A,     // M x K bf16 (shared across batch)
    const u16*  __restrict__ Bt,    // batch x (N x K) bf16
    float*      __restrict__ C,     // batch x (M x N) f32, ld = N
    const float* __restrict__ bias, // length M
    int M, int N, int K, long long strideB, long long strideC)
{
    // [buf][16-row/col group][lane][fragment slot]
    __shared__ __align__(16) u16 As2[2][4][32][16];
    __shared__ __align__(16) u16 Bs2[2][4][32][16];

    const int m0 = blockIdx.x * TM;
    const int n0 = blockIdx.y * TN;
    const u16*  Bb = Bt + (long long)blockIdx.z * strideB;
    float*      Cb = C  + (long long)blockIdx.z * strideC;

    const int tid  = threadIdx.x;
    const int lane = tid & 31;
    const int w    = tid >> 5;        // 8 waves: 4 row-tiles x 2 col-pairs
    const int wr_  = w >> 1, wc_ = w & 1;
    const int lo   = lane & 15, hi = lane >> 4;

    // --- staging geometry (constant per thread) ---
    // tile row/col index r (0..63) + 8-element k-chunk cs (0,8,16,24)
    const int r  = tid >> 2;
    const int cs = (tid & 3) * 8;
    // A (16-bit 16x32 layout, ISA 7.12.2): k = (e&7) + 8*hi + 16*(e>=8)
    const int Awr = r >> 4;
    const int Aln = (r & 15) + (((cs >> 3) & 1) << 4);
    const int Ae0 = (cs & 16) ? 8 : 0;
    // B (32x16): k = (e&7) + 8*hi + 16*(e>=8)
    const int Bcg = r >> 4;
    const int Bln = (r & 15) + (((cs >> 3) & 1) << 4);
    const int Be0 = (cs & 16) ? 8 : 0;

    const u16* aS = A  + (size_t)(m0 + r) * K + cs;
    const u16* bS = Bb + (size_t)(n0 + r) * K + cs;

    auto stage = [&](int buf, int k0) {
#if HAVE_ASYNC
        __builtin_amdgcn_global_load_async_to_lds_b128(
            GAS1(aS + k0), LAS3(&As2[buf][Awr][Aln][Ae0]), 0, 0);
        __builtin_amdgcn_global_load_async_to_lds_b128(
            GAS1(bS + k0), LAS3(&Bs2[buf][Bcg][Bln][Be0]), 0, 0);
#else
        *(uint4*)&As2[buf][Awr][Aln][Ae0] = *(const uint4*)(aS + k0);
        *(uint4*)&Bs2[buf][Bcg][Bln][Be0] = *(const uint4*)(bS + k0);
#endif
    };

    v8f acc0 = {}; v8f acc1 = {};
    const int nk = K / TK;
    stage(0, 0);
    for (int kt = 0; kt < nk; ++kt) {
        wait_async_all();
        __syncthreads();                       // buf (kt&1) fully staged, all readers of other buf done
        if (kt + 1 < nk) stage((kt + 1) & 1, (kt + 1) * TK);   // overlap next copy with WMMA

        const int buf = kt & 1;
        const v8bf* pa  = (const v8bf*)As2[buf][wr_][lane];
        const v8bf* pb0 = (const v8bf*)Bs2[buf][wc_ * 2][lane];
        const v8bf* pb1 = (const v8bf*)Bs2[buf][wc_ * 2 + 1][lane];
        v16bf aF  = __builtin_shufflevector(pa[0],  pa[1],  0,1,2,3,4,5,6,7,8,9,10,11,12,13,14,15);
        v16bf bF0 = __builtin_shufflevector(pb0[0], pb0[1], 0,1,2,3,4,5,6,7,8,9,10,11,12,13,14,15);
        v16bf bF1 = __builtin_shufflevector(pb1[0], pb1[1], 0,1,2,3,4,5,6,7,8,9,10,11,12,13,14,15);

        acc0 = __builtin_amdgcn_wmma_f32_16x16x32_bf16(false, aF, false, bF0,
                                                       (short)0, acc0, false, false);
        acc1 = __builtin_amdgcn_wmma_f32_16x16x32_bf16(false, aF, false, bF1,
                                                       (short)0, acc1, false, false);
    }
#pragma unroll
    for (int e = 0; e < 8; ++e) {              // C layout: M = e + 8*hi, N = lo
        int row = m0 + wr_ * 16 + hi * 8 + e;
        float bv = bias[row];
        size_t base = (size_t)row * N + n0 + wc_ * 32 + lo;
        Cb[base]      = acc0[e] + bv;
        Cb[base + 16] = acc1[e] + bv;
    }
}

// ---------- S4D recurrence + Dp skip + exact GELU -> bf16, emitted transposed ----------
// block = 8 waves = 8 consecutive channels of one batch; lane = state index n.
// Output yT is (B, L, H) so the GEMM can stage it as (N,K) 16B chunks.
__global__ __launch_bounds__(256) void s4d_kernel(
    const float*  __restrict__ hbuf,  // (B,H,L)
    const float4* __restrict__ par,   // (H,N2) for this layer
    const float*  __restrict__ Dp,    // (H) for this layer
    u16*          __restrict__ yT)    // (B,L,H) bf16
{
    __shared__ u16 sh[8][33];
    const int w = threadIdx.x >> 5, lane = threadIdx.x & 31;
    const int b  = blockIdx.x >> 5;          // HC/8 = 32 channel-groups per batch
    const int hg = blockIdx.x & 31;
    const int hh = hg * 8 + w;

    float4 p  = par[hh * N2C + lane];
    float ctr = p.x, cti = p.y, wr = p.z, wi = p.w;
    float dp  = Dp[hh];

    const float* u = hbuf + ((size_t)b * HC + hh) * (size_t)LEN;
    u16* ybase = yT + ((size_t)b * LEN) * HC + hg * 8;
    const int tl = threadIdx.x >> 3, tc = threadIdx.x & 7;

    float sr = 0.f, si = 0.f;
    for (int l0 = 0; l0 < LEN; l0 += 32) {
        float uc   = u[l0 + lane];          // coalesced chunk of 32 timesteps
        float ysel = 0.f;
#pragma unroll
        for (int j = 0; j < 32; ++j) {
            float uj  = __shfl(uc, j, 32);          // broadcast u[l0+j]
            float nsr = wr * sr - wi * si + uj;     // s = w*s + u
            float nsi = wr * si + wi * sr;
            sr = nsr; si = nsi;
            float c = ctr * sr - cti * si;          // Re(Ct * s)
            c += __shfl_xor(c, 16, 32);
            c += __shfl_xor(c,  8, 32);
            c += __shfl_xor(c,  4, 32);
            c += __shfl_xor(c,  2, 32);
            c += __shfl_xor(c,  1, 32);
            if (lane == j) ysel = 2.0f * c;
        }
        float v = ysel + uc * dp;                               // + D*u
        float g = 0.5f * v * (1.0f + erff(v * 0.70710678118654752f)); // exact GELU
        __syncthreads();                    // previous chunk's readers done
        sh[w][lane] = f2bf(g);
        __syncthreads();
        ybase[(size_t)(l0 + tl) * HC + tc] = sh[tc][tl];        // (L,H) layout
    }
}

// ---------- GLU + residual + channel LayerNorm (in-place update of h) ----------
__global__ __launch_bounds__(256) void glu_ln_kernel(
    const float* __restrict__ z,      // (B,2H,L)
    float*       __restrict__ hbuf,   // (B,H,L) in/out
    const float* __restrict__ lnw, const float* __restrict__ lnb)
{
    __shared__ float ssum[256], ssq[256];
    int bl = blockIdx.x;
    int b = bl / LEN, l = bl % LEN;
    int t = threadIdx.x;                       // channel
    size_t zi = (((size_t)b * 2 * HC) + t) * LEN + l;
    float a = z[zi];
    float g = z[zi + (size_t)HC * LEN];
    size_t hix = (((size_t)b * HC) + t) * LEN + l;
    float v = a * (1.0f / (1.0f + expf(-g))) + hbuf[hix];   // GLU + residual
    ssum[t] = v; ssq[t] = v * v;
    __syncthreads();
    for (int s = 128; s > 0; s >>= 1) {
        if (t < s) { ssum[t] += ssum[t + s]; ssq[t] += ssq[t + s]; }
        __syncthreads();
    }
    float m   = ssum[0] * (1.0f / HC);
    float var = ssq[0] * (1.0f / HC) - m * m;
    hbuf[hix] = (v - m) * rsqrtf(var + LN_EPS) * lnw[t] + lnb[t];
}

// ---------- take last timestep ----------
__global__ void tail_kernel(const float* __restrict__ hbuf, float* __restrict__ hl) {
    int i = blockIdx.x * blockDim.x + threadIdx.x;
    if (i < BN * HC) hl[i] = hbuf[(size_t)i * LEN + (LEN - 1)];
}

// ---------- small dense head ----------
__global__ void dense_kernel(const float* __restrict__ in, const float* __restrict__ W,
                             const float* __restrict__ bias, float* __restrict__ out,
                             int Bn, int O, int K, int relu) {
    int i = blockIdx.x * blockDim.x + threadIdx.x;
    if (i >= Bn * O) return;
    int b = i / O, o = i % O;
    const float* xv = in + (size_t)b * K;
    const float* wv = W  + (size_t)o * K;
    float acc = bias[o];
    for (int k = 0; k < K; ++k) acc += xv[k] * wv[k];
    if (relu) acc = fmaxf(acc, 0.f);
    out[i] = acc;
}

// =====================================================================
extern "C" void kernel_launch(void* const* d_in, const int* in_sizes, int n_in,
                              void* d_out, int out_size, void* d_ws, size_t ws_size,
                              hipStream_t stream) {
    const float* x          = (const float*)d_in[0];
    const float* enc_w      = (const float*)d_in[1];
    const float* enc_b      = (const float*)d_in[2];
    const float* log_dt     = (const float*)d_in[3];
    const float* C_re       = (const float*)d_in[4];
    const float* C_im       = (const float*)d_in[5];
    const float* log_A_real = (const float*)d_in[6];
    const float* A_imag     = (const float*)d_in[7];
    const float* Dp         = (const float*)d_in[8];
    const float* out_w      = (const float*)d_in[9];
    const float* out_b      = (const float*)d_in[10];
    const float* ln_w       = (const float*)d_in[11];
    const float* ln_b       = (const float*)d_in[12];
    const float* lin1_w     = (const float*)d_in[13];
    const float* lin1_b     = (const float*)d_in[14];
    const float* lin2_w     = (const float*)d_in[15];
    const float* lin2_b     = (const float*)d_in[16];
    const float* lin3_w     = (const float*)d_in[17];
    const float* lin3_b     = (const float*)d_in[18];
    float* outp = (float*)d_out;

    char* ws = (char*)d_ws;
    size_t off = 0;
    auto alloc = [&](size_t bytes) -> void* {
        void* p = ws + off;
        off = (off + bytes + 255) & ~(size_t)255;
        return p;
    };
    float*  h_buf   = (float*) alloc((size_t)BN * HC * LEN * 4);        // 64 MB
    float*  z_buf   = (float*) alloc((size_t)BN * 2 * HC * LEN * 4);    // 128 MB
    u16*    yT      = (u16*)   alloc((size_t)BN * LEN * HC * 2);        // 32 MB (L,H)
    u16*    xbf     = (u16*)   alloc((size_t)BN * LEN * D_IN * 2);      // 8 MB (L,K)
    u16*    encwbf  = (u16*)   alloc((size_t)HC * D_IN * 2);
    u16*    outwbf  = (u16*)   alloc((size_t)NLAY * 2 * HC * HC * 2);
    float4* par     = (float4*)alloc((size_t)NLAY * HC * N2C * 16);
    float*  hl      = (float*) alloc((size_t)BN * HC * 4);
    float*  t1      = (float*) alloc((size_t)BN * L1C * 4);
    float*  t2      = (float*) alloc((size_t)BN * L2C * 4);
    (void)ws_size; (void)n_in; (void)in_sizes; (void)out_size;

    // weight / input conversions + S4D params
    {
        int n = HC * D_IN;
        f2bf_kernel<<<(n + 255) / 256, 256, 0, stream>>>(enc_w, encwbf, n);
    }
    {
        int n = NLAY * 2 * HC * HC;
        f2bf_kernel<<<(n + 255) / 256, 256, 0, stream>>>(out_w, outwbf, n);
    }
    {
        int n = BN * LEN * D_IN;   // x is already (L, K) per batch for the Bt operand
        f2bf_kernel<<<(n + 255) / 256, 256, 0, stream>>>(x, xbf, n);
    }
    {
        int n = NLAY * HC * N2C;
        s4_params_kernel<<<(n + 255) / 256, 256, 0, stream>>>(
            log_dt, C_re, C_im, log_A_real, A_imag, par);
    }

    // encoder: h[b] = enc_w(256x64) * xbf[b](2048x64)^T + enc_b
    gemm_bf16_kernel<<<dim3(HC / TM, LEN / TN, BN), 256, 0, stream>>>(
        encwbf, xbf, h_buf, enc_b, HC, LEN, D_IN,
        (long long)LEN * D_IN, (long long)HC * LEN);

    for (int layer = 0; layer < NLAY; ++layer) {
        // S4D recurrence + GELU -> yT (B,L,H)
        s4d_kernel<<<BN * (HC / 8), 256, 0, stream>>>(
            h_buf, par + (size_t)layer * HC * N2C, Dp + layer * HC, yT);

        // out-projection: z[b] = out_w[layer](512x256) * yT[b](2048x256)^T + out_b[layer]
        gemm_bf16_kernel<<<dim3(2 * HC / TM, LEN / TN, BN), 256, 0, stream>>>(
            outwbf + (size_t)layer * 2 * HC * HC, yT, z_buf, out_b + layer * 2 * HC,
            2 * HC, LEN, HC, (long long)LEN * HC, (long long)2 * HC * LEN);

        // GLU + residual + LayerNorm (channel), in-place on h
        glu_ln_kernel<<<BN * LEN, 256, 0, stream>>>(
            z_buf, h_buf, ln_w + layer * HC, ln_b + layer * HC);
    }

    // head
    tail_kernel<<<(BN * HC + 255) / 256, 256, 0, stream>>>(h_buf, hl);
    dense_kernel<<<(BN * L1C + 255) / 256, 256, 0, stream>>>(hl, lin1_w, lin1_b, t1, BN, L1C, HC, 1);
    dense_kernel<<<(BN * L2C + 255) / 256, 256, 0, stream>>>(t1, lin2_w, lin2_b, t2, BN, L2C, L1C, 1);
    dense_kernel<<<(BN * DOUT + 255) / 256, 256, 0, stream>>>(t2, lin3_w, lin3_b, outp, BN, DOUT, L2C, 0);
}
